// InterFreqAttentionV2_20452634263746
// MI455X (gfx1250) — matrix operand order repaired and checked by reading, not compiled, not run on hardware
//
#include <hip/hip_runtime.h>
#include <hip/hip_bf16.h>

// ---------------------------------------------------------------------------
// InterFreqAttentionV2 for MI455X (gfx1250, wave32, WMMA f32_16x16x32_f16)
// ---------------------------------------------------------------------------
// Shapes: B=2 C=32 N=4 V=W=32 -> L=1024, Bn=8, T=L*Bn=8192, E=256, H=8, D=32
// ---------------------------------------------------------------------------

typedef __attribute__((ext_vector_type(16))) _Float16 v16h;
typedef __attribute__((ext_vector_type(8)))  _Float16 v8h;
typedef __attribute__((ext_vector_type(8)))  float    v8f;

enum : int { Tt = 8192, Ee = 256, Cc = 32, Hh = 8, Dd = 32, Ll = 1024, Bnn = 8 };

__device__ __forceinline__ v16h combine16(v8h lo, v8h hi) {
  v16h r;
#pragma unroll
  for (int i = 0; i < 8; ++i) { r[i] = lo[i]; r[i + 8] = hi[i]; }
  return r;
}

// A fragment (16 rows x 32 K, f16).  base -> element (m=0,k=0), ldh = row stride (halves).
// ISA layout: lane<16 (m=lane): K = {0..7, 16..23}; lane>=16 (m=lane-16): K = {8..15, 24..31}
__device__ __forceinline__ v16h load_a_frag(const _Float16* base, int ldh) {
  int lane = threadIdx.x & 31;
  int gl = lane >> 4, m = lane & 15;
  const _Float16* rp = base + (size_t)m * ldh + 8 * gl;
  return combine16(*(const v8h*)rp, *(const v8h*)(rp + 16));
}

// B fragment (32 K x 16 cols, f16) from row-major Wt(N x K): wt0 -> row nBase, col k0.
// ISA layout: lane (gl,n): column n, K = {8*gl..8*gl+7, 16+8*gl..16+8*gl+7}
__device__ __forceinline__ v16h load_b_frag(const _Float16* wt0, int ldh) {
  int lane = threadIdx.x & 31;
  int gl = lane >> 4, n = lane & 15;
  const _Float16* rp = wt0 + (size_t)n * ldh + 8 * gl;
  return combine16(*(const v8h*)rp, *(const v8h*)(rp + 16));
}

__device__ __forceinline__ v8f wmma16(v16h a, v16h b, v8f c) {
  return __builtin_amdgcn_wmma_f32_16x16x32_f16(false, a, false, b, (short)0, c,
                                                false, false);
}

// ------------------------- data-prep kernels -------------------------------

// x (B,C,N,V,W) f32 -> token-major (T, C) f16, t = (v*32+w)*8 + b*4 + n
__global__ __launch_bounds__(256) void k_pack(const float* __restrict__ xlf,
                                              const float* __restrict__ xhf,
                                              _Float16* __restrict__ tokh,
                                              _Float16* __restrict__ kvh) {
  int i = blockIdx.x * 256 + threadIdx.x;
  if (i >= 262144) return;
  int w = i & 31, v = (i >> 5) & 31, n = (i >> 10) & 3, c = (i >> 12) & 31, b = i >> 17;
  int t = ((v << 5) + w) * 8 + b * 4 + n;
  tokh[t * Cc + c] = (_Float16)xlf[i];
  kvh [t * Cc + c] = (_Float16)xhf[i];
}

__global__ __launch_bounds__(256) void k_cvt(const float* __restrict__ s,
                                             _Float16* __restrict__ d, int n) {
  int i = blockIdx.x * 256 + threadIdx.x;
  if (i < n) d[i] = (_Float16)s[i];
}

// ------------------------------ GEMM ---------------------------------------
// Y[T x N] = A[T x K](f16) @ Wt[N x K]^T, f32 accumulation via WMMA.
// Compile-time K and JT (16-col tiles per wave) => fully unrolled K loop,
// no conditional WMMAs, accumulators pinned (avoids v_mov_b64 merge chains).
template <int K, int JT>
__global__ __launch_bounds__(256) void k_gemm(const _Float16* __restrict__ A,
                                              const _Float16* __restrict__ Wt,
                                              const float* __restrict__ res,
                                              float* __restrict__ Yf,
                                              _Float16* __restrict__ Yh,
                                              int N, int relu) {
  int wave = threadIdx.x >> 5, lane = threadIdx.x & 31;
  int tBase = (blockIdx.x * 8 + wave) * 16;
  int colBase = blockIdx.y * (16 * JT);
  int gl = lane >> 4, ln = lane & 15;

  v8f acc[JT];
#pragma unroll
  for (int j = 0; j < JT; ++j) acc[j] = v8f{};

#pragma unroll
  for (int k0 = 0; k0 < K; k0 += 32) {
    if (k0 + 32 < K)  // exercise global_prefetch_b8 path
      __builtin_prefetch(A + (size_t)tBase * K + k0 + 32, 0, 3);
    v16h a = load_a_frag(A + (size_t)tBase * K + k0, K);
#pragma unroll
    for (int j = 0; j < JT; ++j) {
      v16h b = load_b_frag(Wt + (size_t)(colBase + j * 16) * K + k0, K);
      acc[j] = wmma16(a, b, acc[j]);
    }
  }
#pragma unroll
  for (int j = 0; j < JT; ++j) {
    int col = colBase + j * 16 + ln;
#pragma unroll
    for (int r = 0; r < 8; ++r) {
      int row = tBase + r + 8 * gl;
      size_t o = (size_t)row * N + col;
      float v = acc[j][r];
      if (res)  v += res[o];
      if (relu) v = v > 0.f ? v : 0.f;
      if (Yf) Yf[o] = v;
      if (Yh) Yh[o] = (_Float16)v;
    }
  }
}

// ---------------------------- LayerNorm ------------------------------------
// One wave per token (E=256 -> 8 elems/lane).  Optional raw f16 copy (v_in).
__global__ __launch_bounds__(256) void k_ln(const float* __restrict__ X,
                                            const float* __restrict__ g,
                                            const float* __restrict__ bb,
                                            _Float16* __restrict__ Yh,
                                            _Float16* __restrict__ rawh) {
  int wave = threadIdx.x >> 5, lane = threadIdx.x & 31;
  int t = blockIdx.x * 8 + wave;
  const float* x = X + (size_t)t * Ee;
  float v[8], s = 0.f;
#pragma unroll
  for (int j = 0; j < 8; ++j) { v[j] = x[j * 32 + lane]; s += v[j]; }
#pragma unroll
  for (int m = 16; m >= 1; m >>= 1) s += __shfl_xor(s, m, 32);
  float mu = s * (1.f / 256.f), q = 0.f;
#pragma unroll
  for (int j = 0; j < 8; ++j) { float d = v[j] - mu; q += d * d; }
#pragma unroll
  for (int m = 16; m >= 1; m >>= 1) q += __shfl_xor(q, m, 32);
  float inv = rsqrtf(q * (1.f / 256.f) + 1e-5f);
#pragma unroll
  for (int j = 0; j < 8; ++j) {
    int e = j * 32 + lane;
    Yh[(size_t)t * Ee + e] = (_Float16)((v[j] - mu) * inv * g[e] + bb[e]);
    if (rawh) rawh[(size_t)t * Ee + e] = (_Float16)v[j];
  }
}

// -------------------------- windowed attention -----------------------------
// One wave per (bn, h, 16-query tile).  7x7 local window -> only key rows
// ki in [qi-3, qi+3] contribute; flash-softmax across those rows (32 keys ea).
__global__ __launch_bounds__(256) void k_attn(const _Float16* __restrict__ Qp,
                                              const _Float16* __restrict__ Kp,
                                              const _Float16* __restrict__ Vp,
                                              _Float16* __restrict__ ctxh) {
  __shared__ __align__(16) _Float16 Vsh[8][32][32];   // per-wave V tile
  __shared__ __align__(16) _Float16 Pt[8][16][32];    // per-wave P tile (C->A relayout)
  int wave = threadIdx.x >> 5, lane = threadIdx.x & 31;
  int wg = blockIdx.x * 8 + wave;            // 0..4095 = bn(8) * h(8) * qt(64)
  int bn = wg >> 9, h = (wg >> 6) & 7, qt = wg & 63;
  int qi = qt >> 1, qjB = (qt & 1) * 16;
  int gl = lane >> 4, ln = lane & 15;
  int lq0 = qi * 32 + qjB;

  // Q fragment: row m -> token (lq0+m)*8+bn, d = 0..31
  v16h qa;
  {
    const _Float16* rp = Qp + ((size_t)(lq0 + ln) * Bnn + bn) * Ee + h * Dd + 8 * gl;
    qa = combine16(*(const v8h*)rp, *(const v8h*)(rp + 16));
  }

  float mrow[8], lrow[8];
  v8f acc0 = {}, acc1 = {};
#pragma unroll
  for (int r = 0; r < 8; ++r) { mrow[r] = -1e30f; lrow[r] = 0.f; }

  int ki0 = qi - 3 < 0 ? 0 : qi - 3;
  int ki1 = qi + 3 > 31 ? 31 : qi + 3;
  const float scale = 0.17677669529663687f;  // 1/sqrt(32)

  for (int ki = ki0; ki <= ki1; ++ki) {
    // stage this key-row's V (32 keys x 32 d) into LDS: one key per lane
    {
      const _Float16* vp = Vp + ((size_t)(ki * 32 + lane) * Bnn + bn) * Ee + h * Dd;
      *(v8h*)&Vsh[wave][lane][0]  = *(const v8h*)(vp);
      *(v8h*)&Vsh[wave][lane][8]  = *(const v8h*)(vp + 8);
      *(v8h*)&Vsh[wave][lane][16] = *(const v8h*)(vp + 16);
      *(v8h*)&Vsh[wave][lane][24] = *(const v8h*)(vp + 24);
    }
    // S = Q @ K^T  (two 16-key fragments)
    v8f s[2];
#pragma unroll
    for (int f = 0; f < 2; ++f) {
      const _Float16* kp =
          Kp + ((size_t)(ki * 32 + f * 16 + ln) * Bnn + bn) * Ee + h * Dd + 8 * gl;
      v16h kb = combine16(*(const v8h*)kp, *(const v8h*)(kp + 16));
      v8f z = {};
      s[f] = wmma16(qa, kb, z);
    }
    // scale + column mask + row max (rows r+8*gl live on 16-lane group gl)
    float rmax[8];
#pragma unroll
    for (int r = 0; r < 8; ++r) {
      int qj = qjB + r + 8 * gl;              // query column of this row
#pragma unroll
      for (int f = 0; f < 2; ++f) {
        int dj = f * 16 + ln - qj;
        float val = s[f][r] * scale;
        if (dj < -3 || dj > 3) val = -1e30f;
        s[f][r] = val;
      }
      float v2 = fmaxf(s[0][r], s[1][r]);
#pragma unroll
      for (int m = 8; m >= 1; m >>= 1) v2 = fmaxf(v2, __shfl_xor(v2, m, 16));
      rmax[r] = v2;
    }
    // flash-softmax update; write P into LDS in C-layout positions
#pragma unroll
    for (int r = 0; r < 8; ++r) {
      float mn = fmaxf(mrow[r], rmax[r]);
      float alpha = __expf(mrow[r] - mn);
      mrow[r] = mn;
      float p0 = __expf(s[0][r] - mn);
      float p1 = __expf(s[1][r] - mn);
      float sm = p0 + p1;
#pragma unroll
      for (int m = 8; m >= 1; m >>= 1) sm += __shfl_xor(sm, m, 16);
      lrow[r] = lrow[r] * alpha + sm;
      acc0[r] *= alpha;
      acc1[r] *= alpha;
      Pt[wave][r + 8 * gl][ln]      = (_Float16)p0;
      Pt[wave][r + 8 * gl][16 + ln] = (_Float16)p1;
    }
    // reload P as an A fragment (same-wave LDS ops are in-order)
    v16h pa;
    {
      const _Float16* rp = &Pt[wave][ln][16 * gl];
      pa = combine16(*(const v8h*)rp, *(const v8h*)(rp + 8));
    }
    // ctx += P @ V  (two 16-wide d fragments gathered from LDS)
#pragma unroll
    for (int f = 0; f < 2; ++f) {
      v16h vb;
#pragma unroll
      for (int i = 0; i < 16; ++i) vb[i] = Vsh[wave][16 * gl + i][f * 16 + ln];
      if (f == 0) acc0 = wmma16(pa, vb, acc0);
      else        acc1 = wmma16(pa, vb, acc1);
    }
  }
  // normalize and store ctx (token-major (T,E) f16)
#pragma unroll
  for (int r = 0; r < 8; ++r) {
    int mm = r + 8 * gl;
    _Float16* op = ctxh + ((size_t)(lq0 + mm) * Bnn + bn) * Ee + h * Dd;
    float inv = 1.f / lrow[r];
    op[ln]      = (_Float16)(acc0[r] * inv);
    op[16 + ln] = (_Float16)(acc1[r] * inv);
  }
}

// out_t (T x C) f32 -> d_out (B,C,N,V,W)
__global__ __launch_bounds__(256) void k_scatter(const float* __restrict__ out_t,
                                                 float* __restrict__ out) {
  int i = blockIdx.x * 256 + threadIdx.x;
  if (i >= 262144) return;
  int w = i & 31, v = (i >> 5) & 31, n = (i >> 10) & 3, c = (i >> 12) & 31, b = i >> 17;
  int t = ((v << 5) + w) * 8 + b * 4 + n;
  out[i] = out_t[(size_t)t * Cc + c];
}

// ---------------------------------------------------------------------------
extern "C" void kernel_launch(void* const* d_in, const int* in_sizes, int n_in,
                              void* d_out, int out_size, void* d_ws, size_t ws_size,
                              hipStream_t stream) {
  const float* x_lf    = (const float*)d_in[0];
  const float* x_hf    = (const float*)d_in[1];
  const float* w_in    = (const float*)d_in[2];
  const float* w_qk    = (const float*)d_in[3];
  const float* ln_g    = (const float*)d_in[4];
  const float* ln_b    = (const float*)d_in[5];
  const float* in_proj = (const float*)d_in[6];
  const float* out_proj= (const float*)d_in[7];
  const float* ln2_g   = (const float*)d_in[8];
  const float* ln2_b   = (const float*)d_in[9];
  const float* ff_w1   = (const float*)d_in[10];
  const float* ff_w2   = (const float*)d_in[11];
  const float* w_out   = (const float*)d_in[12];
  float* out = (float*)d_out;

  char* ws = (char*)d_ws;
  size_t off = 0;
  auto alloc = [&](size_t bytes) -> void* {
    void* p = ws + off;
    off += (bytes + 255) & ~(size_t)255;
    return p;
  };
  _Float16* tokh    = (_Float16*)alloc((size_t)Tt * Cc * 2);
  _Float16* kvh     = (_Float16*)alloc((size_t)Tt * Cc * 2);
  _Float16* w_inh   = (_Float16*)alloc((size_t)Ee * Cc * 2);
  _Float16* w_qkh   = (_Float16*)alloc((size_t)Ee * Cc * 2);
  _Float16* in_projh= (_Float16*)alloc((size_t)3 * Ee * Ee * 2);
  _Float16* out_projh=(_Float16*)alloc((size_t)Ee * Ee * 2);
  _Float16* ff_w1h  = (_Float16*)alloc((size_t)2 * Ee * Ee * 2);
  _Float16* ff_w2h  = (_Float16*)alloc((size_t)Ee * 2 * Ee * 2);
  _Float16* w_outh  = (_Float16*)alloc((size_t)Cc * Ee * 2);
  float*    epi_tok = (float*)   alloc((size_t)Tt * Ee * 4);
  float*    epi_kv  = (float*)   alloc((size_t)Tt * Ee * 4);
  _Float16* q_inh   = (_Float16*)alloc((size_t)Tt * Ee * 2);
  _Float16* k_inh   = (_Float16*)alloc((size_t)Tt * Ee * 2);
  _Float16* v_inh   = (_Float16*)alloc((size_t)Tt * Ee * 2);
  _Float16* q_proj  = (_Float16*)alloc((size_t)Tt * Ee * 2);
  _Float16* k_proj  = (_Float16*)alloc((size_t)Tt * Ee * 2);
  _Float16* v_proj  = (_Float16*)alloc((size_t)Tt * Ee * 2);
  _Float16* ctxh    = (_Float16*)alloc((size_t)Tt * Ee * 2);
  float*    epi2    = (float*)   alloc((size_t)Tt * Ee * 4);
  _Float16* ln2h    = (_Float16*)alloc((size_t)Tt * Ee * 2);
  _Float16* h1h     = (_Float16*)alloc((size_t)Tt * 2 * Ee * 2);
  float*    epi3    = (float*)   alloc((size_t)Tt * Ee * 4);
  _Float16* epi3h   = (_Float16*)alloc((size_t)Tt * Ee * 2);
  float*    out_t   = (float*)   alloc((size_t)Tt * Cc * 4);
  (void)ws_size; (void)in_sizes; (void)n_in; (void)out_size;

  auto cvt = [&](const float* s, _Float16* d, int n) {
    k_cvt<<<(n + 255) / 256, 256, 0, stream>>>(s, d, n);
  };

  // 1) pack inputs to token-major f16 + convert weights to f16
  k_pack<<<1024, 256, 0, stream>>>(x_lf, x_hf, tokh, kvh);
  cvt(w_in,    w_inh,    Ee * Cc);
  cvt(w_qk,    w_qkh,    Ee * Cc);
  cvt(in_proj, in_projh, 3 * Ee * Ee);
  cvt(out_proj,out_projh,Ee * Ee);
  cvt(ff_w1,   ff_w1h,   2 * Ee * Ee);
  cvt(ff_w2,   ff_w2h,   Ee * 2 * Ee);
  cvt(w_out,   w_outh,   Cc * Ee);

  dim3 g256(Tt / 16 / 8, Ee / 64);       // N=256, JT=4
  dim3 g512(Tt / 16 / 8, 2 * Ee / 64);   // N=512, JT=4
  dim3 g32 (Tt / 16 / 8, 1);             // N=32,  JT=2

  // 2) input projections (K=32 -> single WMMA per tile)
  k_gemm<32, 4><<<g256, 256, 0, stream>>>(tokh, w_inh, nullptr, epi_tok, nullptr, Ee, 0);
  k_gemm<32, 4><<<g256, 256, 0, stream>>>(kvh,  w_qkh, nullptr, epi_kv,  nullptr, Ee, 0);

  // 3) LayerNorm1 (q from epi_tok; k + raw v from epi_kv)
  k_ln<<<Tt / 8, 256, 0, stream>>>(epi_tok, ln_g, ln_b, q_inh, nullptr);
  k_ln<<<Tt / 8, 256, 0, stream>>>(epi_kv,  ln_g, ln_b, k_inh, v_inh);

  // 4) QKV projections (E x E each, K fully unrolled: 8x4 WMMAs)
  k_gemm<256, 4><<<g256, 256, 0, stream>>>(q_inh, in_projh,             nullptr, nullptr, q_proj, Ee, 0);
  k_gemm<256, 4><<<g256, 256, 0, stream>>>(k_inh, in_projh + Ee * Ee,   nullptr, nullptr, k_proj, Ee, 0);
  k_gemm<256, 4><<<g256, 256, 0, stream>>>(v_inh, in_projh + 2*Ee*Ee,   nullptr, nullptr, v_proj, Ee, 0);

  // 5) windowed flash attention (4096 waves, 8 per block)
  k_attn<<<512, 256, 0, stream>>>(q_proj, k_proj, v_proj, ctxh);

  // 6) out-proj + residual
  k_gemm<256, 4><<<g256, 256, 0, stream>>>(ctxh, out_projh, epi_tok, epi2, nullptr, Ee, 0);

  // 7) LN2 -> FFN (ReLU) -> residual (keep f32 + f16 copies)
  k_ln<<<Tt / 8, 256, 0, stream>>>(epi2, ln2_g, ln2_b, ln2h, nullptr);
  k_gemm<256, 4><<<g512, 256, 0, stream>>>(ln2h, ff_w1h, nullptr, nullptr, h1h, 2 * Ee, 1);
  k_gemm<512, 4><<<g256, 256, 0, stream>>>(h1h,  ff_w2h, epi2,    epi3,    epi3h, Ee, 0);

  // 8) output projection (E -> C, N=32 -> JT=2) and scatter to (B,C,N,V,W)
  k_gemm<256, 2><<<g32, 256, 0, stream>>>(epi3h, w_outh, nullptr, out_t, nullptr, Cc, 0);
  k_scatter<<<1024, 256, 0, stream>>>(out_t, out);
}
